// GCN_59639915872756
// MI455X (gfx1250) — matrix operand order repaired and checked by reading, not compiled, not run on hardware
//
#include <hip/hip_runtime.h>
#include <stdint.h>

typedef __attribute__((ext_vector_type(16))) _Float16 v16h;
typedef __attribute__((ext_vector_type(8)))  _Float16 v8h;
typedef __attribute__((ext_vector_type(4)))  _Float16 v4h;
typedef __attribute__((ext_vector_type(8)))  float    v8f;

#define N_NODES   50000
#define N_EDGES   800000
#define NUM_REL   8
#define NUM_BASES 30
#define DIM       128

// ---------------------------------------------------------------------------
// W[r,i,o] = sum_b comp[r,b] * basis[b,i,o]
// ---------------------------------------------------------------------------
__global__ void compW_kernel(const float* __restrict__ comp,
                             const float* __restrict__ basis,
                             float* __restrict__ W) {
    int idx = blockIdx.x * blockDim.x + threadIdx.x;   // r*16384 + i*128 + o
    if (idx >= NUM_REL * DIM * DIM) return;
    int r  = idx >> 14;
    int io = idx & 16383;
    float s = 0.0f;
#pragma unroll
    for (int b = 0; b < NUM_BASES; ++b)
        s += comp[r * NUM_BASES + b] * basis[b * DIM * DIM + io];
    W[idx] = s;
}

// ---------------------------------------------------------------------------
// Pack fp32 [128 x 128] (row=K, col=N) weight matrices into the wave32 WMMA
// B-fragment order used here for v_wmma_f32_16x16x32_f16 (contiguous k, same as A):
//   per k-tile (K=32) / n-tile (N=16): lane holds column N = ntile*16+(lane&15),
//   16 contiguous halves covering K = ktile*32 + (lane<16 ? 0 : 16) + i.
// Packed index = (((m*4 + kt)*8 + nt)*32 + lane)*16 + i
// ---------------------------------------------------------------------------
__global__ void packB_kernel(const float* __restrict__ B,
                             _Float16* __restrict__ P, int nmat) {
    int t = blockIdx.x * blockDim.x + threadIdx.x;
    if (t >= nmat * 4 * 8 * 32 * 16) return;
    int i    = t & 15;
    int lane = (t >> 4) & 31;
    int nt   = (t >> 9) & 7;
    int kt   = (t >> 12) & 3;
    int m    = t >> 14;
    int K = kt * 32 + ((lane < 16) ? 0 : 16) + i;
    int N = nt * 16 + (lane & 15);
    P[t] = (_Float16)B[m * DIM * DIM + K * DIM + N];
}

// f32 -> f16, 4 elements per thread (b128 load / b64 store)
__global__ void cvt_f16_kernel(const float* __restrict__ x,
                               _Float16* __restrict__ y, int n4) {
    int t = blockIdx.x * blockDim.x + threadIdx.x;
    if (t >= n4) return;
    const float4 v = ((const float4*)x)[t];
    v4h o;
    o[0] = (_Float16)v.x; o[1] = (_Float16)v.y;
    o[2] = (_Float16)v.z; o[3] = (_Float16)v.w;
    ((v4h*)y)[t] = o;
}

// zero fp32, 4 elements per thread (b128 store)
__global__ void zero_f32_kernel(float* __restrict__ x, int n4) {
    int t = blockIdx.x * blockDim.x + threadIdx.x;
    if (t >= n4) return;
    ((float4*)x)[t] = make_float4(0.f, 0.f, 0.f, 0.f);
}

// ---------------------------------------------------------------------------
// One K=128 WMMA pass: acc += A[row, :] tile  @  B (pre-packed fragments).
// Direct pointer use (no selects) so LLVM infers addrspace(1) -> global_load.
// ---------------------------------------------------------------------------
__device__ __forceinline__ v8f wmma_pass(const _Float16* __restrict__ A,
                                         const _Float16* __restrict__ BP,
                                         v8f acc, int row, int wave, int lane) {
    const int aoff = (lane < 16) ? 0 : 16;      // A-fragment K sub-offset
    const _Float16* Ap = A + (size_t)row * DIM + aoff;
    const _Float16* Bp = BP + ((size_t)(wave * 32 + lane) << 4);
#pragma unroll
    for (int kt = 0; kt < 4; ++kt) {
        // A fragment: lane<16 holds K = 32kt+{0..15} of this row;
        // lane>=16 holds K = 32kt+{16..31}. Two contiguous 16B loads.
        const v8h alo = *(const v8h*)(Ap + kt * 32);
        const v8h ahi = *(const v8h*)(Ap + kt * 32 + 8);
        v16h a;
#pragma unroll
        for (int i = 0; i < 8; ++i) { a[i] = alo[i]; a[i + 8] = ahi[i]; }
        // B fragment: 16 contiguous pre-packed halves per lane per k-tile.
        const v8h blo = *(const v8h*)(Bp + kt * 4096);
        const v8h bhi = *(const v8h*)(Bp + kt * 4096 + 8);
        v16h b;
#pragma unroll
        for (int i = 0; i < 8; ++i) { b[i] = blo[i]; b[i + 8] = bhi[i]; }
        acc = __builtin_amdgcn_wmma_f32_16x16x32_f16(
            false, a, false, b, (short)0, acc, false, false);
    }
    return acc;
}

// ---------------------------------------------------------------------------
// WMMA GEMM:  C[nrows x 128] = A0 @ B0  (+ A1 @ B1 if dual)  (+ bias)
// Block: 256 threads = 8 wave32; wave w owns columns [16w, 16w+16);
// block owns 16 rows; grid = nrows/16 exactly (50000/16 = 3125) so EXEC is
// all-ones everywhere (WMMA requirement). `dual`/`hasBias` are uniform scalar
// branches -> EXEC untouched.
// ---------------------------------------------------------------------------
__global__ __launch_bounds__(256) void gemm128_kernel(
    const _Float16* __restrict__ A0, const _Float16* __restrict__ B0,
    const _Float16* __restrict__ A1, const _Float16* __restrict__ B1,
    const float* __restrict__ bias, float* __restrict__ C,
    int dual, int hasBias)
{
    const int wave = threadIdx.x >> 5;
    const int lane = threadIdx.x & 31;
    const int m0   = blockIdx.x << 4;
    const int mrow = lane & 15;
    const int col  = (wave << 4) + mrow;

    v8f acc;
    const float bv = hasBias ? bias[col] : 0.0f;
#pragma unroll
    for (int j = 0; j < 8; ++j) acc[j] = bv;

    acc = wmma_pass(A0, B0, acc, m0 + mrow, wave, lane);
    if (dual)
        acc = wmma_pass(A1, B1, acc, m0 + mrow, wave, lane);

    // C/D layout: VGPR j -> row m0 + j + (lane<16 ? 0 : 8), col as above.
    float* Cp = C + (size_t)(m0 + ((lane < 16) ? 0 : 8)) * DIM + col;
#pragma unroll
    for (int j = 0; j < 8; ++j) Cp[(size_t)j * DIM] = acc[j];
}

// ---------------------------------------------------------------------------
// Edge pass, layer 1 (relation-filtered):
//   h[dst] += norm[e] * xproj_r[src]   for edges with edge_type == rel
// One wave32 per edge (grid-stride); 4 floats per lane (128 = 32*4).
// xproj_r is 25.6 MB -> L2-resident on MI455X (192 MB global L2): the random
// gathers and the scatter atomics mostly never touch HBM.
// ---------------------------------------------------------------------------
__global__ __launch_bounds__(256) void edge_rgcn_kernel(
    const int* __restrict__ ei, const int* __restrict__ et,
    const float* __restrict__ enorm, const float* __restrict__ xproj,
    float* __restrict__ h, int rel)
{
    const int lane = threadIdx.x & 31;
    const int gw   = (blockIdx.x * blockDim.x + threadIdx.x) >> 5;
    const int nw   = (gridDim.x * blockDim.x) >> 5;
    for (int e = gw; e < N_EDGES; e += nw) {
        if (et[e] != rel) continue;
        const int   src = ei[e];
        const int   dst = ei[N_EDGES + e];
        const float w   = enorm[e];
        const float4 v = *(const float4*)(xproj + (size_t)src * DIM + lane * 4);
        float* hp = h + (size_t)dst * DIM + lane * 4;
        atomicAdd(hp + 0, v.x * w);
        atomicAdd(hp + 1, v.y * w);
        atomicAdd(hp + 2, v.z * w);
        atomicAdd(hp + 3, v.w * w);
    }
}

// Edge pass, layer 2:  agg2[dst] += h[src]
__global__ __launch_bounds__(256) void edge_plain_kernel(
    const int* __restrict__ ei, const float* __restrict__ h,
    float* __restrict__ agg2)
{
    const int lane = threadIdx.x & 31;
    const int gw   = (blockIdx.x * blockDim.x + threadIdx.x) >> 5;
    const int nw   = (gridDim.x * blockDim.x) >> 5;
    for (int e = gw; e < N_EDGES; e += nw) {
        const int src = ei[e];
        const int dst = ei[N_EDGES + e];
        const float4 v = *(const float4*)(h + (size_t)src * DIM + lane * 4);
        float* ap = agg2 + (size_t)dst * DIM + lane * 4;
        atomicAdd(ap + 0, v.x);
        atomicAdd(ap + 1, v.y);
        atomicAdd(ap + 2, v.z);
        atomicAdd(ap + 3, v.w);
    }
}

// ---------------------------------------------------------------------------
extern "C" void kernel_launch(void* const* d_in, const int* in_sizes, int n_in,
                              void* d_out, int out_size, void* d_ws, size_t ws_size,
                              hipStream_t stream) {
    const float* node_feat = (const float*)d_in[0];   // [50000,128]
    const int*   edge_idx  = (const int*)d_in[1];     // [2,800000]
    const int*   edge_type = (const int*)d_in[2];     // [800000]
    const float* edge_norm = (const float*)d_in[3];   // [800000]
    const float* basis     = (const float*)d_in[4];   // [30,128,128]
    const float* comp      = (const float*)d_in[5];   // [8,30]
    const float* root      = (const float*)d_in[6];   // [128,128]
    const float* bias1     = (const float*)d_in[7];   // [128]
    const float* w_rel     = (const float*)d_in[8];   // [128,128]
    const float* b_rel     = (const float*)d_in[9];   // [128]
    const float* w_root    = (const float*)d_in[10];  // [128,128]
    float* out = (float*)d_out;

    // ---- carve workspace (256B aligned slabs) ----
    uintptr_t p = ((uintptr_t)d_ws + 255) & ~(uintptr_t)255;
    auto carve = [&](size_t bytes) -> uintptr_t {
        uintptr_t q = p;
        p = (p + bytes + 255) & ~(uintptr_t)255;
        return q;
    };
    float*    W      = (float*)   carve((size_t)NUM_REL * DIM * DIM * 4);   // 0.5 MB
    _Float16* WP     = (_Float16*)carve((size_t)NUM_REL * DIM * DIM * 2);   // 0.25 MB
    _Float16* rootP  = (_Float16*)carve((size_t)DIM * DIM * 2);
    _Float16* wrelP  = (_Float16*)carve((size_t)DIM * DIM * 2);
    _Float16* wrootP = (_Float16*)carve((size_t)DIM * DIM * 2);
    _Float16* Xh     = (_Float16*)carve((size_t)N_NODES * DIM * 2);         // 12.8 MB
    float*    xproj  = (float*)   carve((size_t)N_NODES * DIM * 4);         // 25.6 MB
    float*    h      = (float*)   carve((size_t)N_NODES * DIM * 4);         // 25.6 MB
    _Float16* hh     = (_Float16*)carve((size_t)N_NODES * DIM * 2);         // 12.8 MB
    float*    agg2   = (float*)   carve((size_t)N_NODES * DIM * 4);         // 25.6 MB
    _Float16* agg2h  = (_Float16*)carve((size_t)N_NODES * DIM * 2);         // 12.8 MB

    const int NF   = N_NODES * DIM;                 // 6,400,000
    const int NF4  = NF / 4;                        // 1,600,000 (vec4 threads)
    const int TB   = 256;
    const int gemmBlocks = N_NODES / 16;            // 3125 (exact)
    const int edgeBlocks = 2048;

    // 1) W = comp x basis, then pack all weights into WMMA B-fragment layout
    compW_kernel<<<(NUM_REL * DIM * DIM + TB - 1) / TB, TB, 0, stream>>>(comp, basis, W);
    packB_kernel<<<(NUM_REL * DIM * DIM + TB - 1) / TB, TB, 0, stream>>>(W, WP, NUM_REL);
    packB_kernel<<<(DIM * DIM + TB - 1) / TB, TB, 0, stream>>>(root,   rootP,  1);
    packB_kernel<<<(DIM * DIM + TB - 1) / TB, TB, 0, stream>>>(w_rel,  wrelP,  1);
    packB_kernel<<<(DIM * DIM + TB - 1) / TB, TB, 0, stream>>>(w_root, wrootP, 1);

    // 2) f16 copy of node features (halves GEMM A traffic)
    cvt_f16_kernel<<<(NF4 + TB - 1) / TB, TB, 0, stream>>>(node_feat, Xh, NF4);

    // 3) h = X @ root + bias1  (WMMA)
    gemm128_kernel<<<gemmBlocks, TB, 0, stream>>>(Xh, rootP, Xh, rootP,
                                                  bias1, h, 0, 1);

    // 4) per-relation: xproj_r = X @ W[r] (WMMA), then L2-resident scatter pass
    for (int r = 0; r < NUM_REL; ++r) {
        gemm128_kernel<<<gemmBlocks, TB, 0, stream>>>(
            Xh, WP + (size_t)r * DIM * DIM, Xh, WP,
            bias1, xproj, 0, 0);
        edge_rgcn_kernel<<<edgeBlocks, TB, 0, stream>>>(
            edge_idx, edge_type, edge_norm, xproj, h, r);
    }

    // 5) layer 2 aggregation: agg2 = segment_sum(h[src] -> dst)
    zero_f32_kernel<<<(NF4 + TB - 1) / TB, TB, 0, stream>>>(agg2, NF4);
    edge_plain_kernel<<<edgeBlocks, TB, 0, stream>>>(edge_idx, h, agg2);

    // 6) out = agg2 @ w_rel + h @ w_root + b_rel  (fused dual WMMA GEMM)
    cvt_f16_kernel<<<(NF4 + TB - 1) / TB, TB, 0, stream>>>(h, hh, NF4);
    cvt_f16_kernel<<<(NF4 + TB - 1) / TB, TB, 0, stream>>>(agg2, agg2h, NF4);
    gemm128_kernel<<<gemmBlocks, TB, 0, stream>>>(agg2h, wrelP, hh, wrootP,
                                                  b_rel, out, 1, 1);
}
